// AlignmentMatrix_9268539425543
// MI455X (gfx1250) — matrix operand order repaired and checked by reading, not compiled, not run on hardware
//
#include <hip/hip_runtime.h>

typedef _Float16 v16h __attribute__((ext_vector_type(16)));
typedef _Float16 v8h  __attribute__((ext_vector_type(8)));
typedef _Float16 v4h  __attribute__((ext_vector_type(4)));
typedef float    v8f  __attribute__((ext_vector_type(8)));

#define H2  128
#define LSEQ 1024
#define BM  128
#define BN  128
#define PAD 8
#define LDK (H2 + PAD)   // 136 halfs = 272B row stride -> conflict-free frag reads

union F16Frag { v16h v; v8h h[2]; };

__global__ __launch_bounds__(256) void alignment_matrix_wmma_kernel(
    const float* __restrict__ body,
    const float* __restrict__ pun,
    const float* __restrict__ w_u,
    float* __restrict__ out)
{
    __shared__ __align__(16) _Float16 sA[BM][LDK];   // body * w3, f16
    __shared__ __align__(16) _Float16 sB[BN][LDK];   // pun, f16
    __shared__ float sBias[BM + BN];                 // s_body rows | s_pun cols

    const int tid  = threadIdx.x;
    const int lane = tid & 31;
    const int wave = tid >> 5;

    const int b  = blockIdx.z;
    const int i0 = blockIdx.y * BM;
    const int j0 = blockIdx.x * BN;

    const float* bodyB = body + ((size_t)b * LSEQ + i0) * H2;
    const float* punB  = pun  + ((size_t)b * LSEQ + j0) * H2;
    const float* w1 = w_u;
    const float* w2 = w_u + H2;
    const float* w3 = w_u + 2 * H2;

    // ---- Stage tiles into LDS as f16 (body pre-scaled by w3) ----
    // 128 rows x 128 cols, 256 threads, 16 float4 each; coalesced 512B/wave.
#pragma unroll
    for (int r = 0; r < 16; ++r) {
        int idx = r * 256 + tid;          // 0..4095
        int row = idx >> 5;               // 0..127
        int c   = (idx & 31) * 4;         // 0..124
        float4 vb  = *(const float4*)(bodyB + row * H2 + c);
        float4 vp  = *(const float4*)(punB  + row * H2 + c);
        float4 w3v = *(const float4*)(w3 + c);
        v4h hb = { (_Float16)(vb.x * w3v.x), (_Float16)(vb.y * w3v.y),
                   (_Float16)(vb.z * w3v.z), (_Float16)(vb.w * w3v.w) };
        v4h hp = { (_Float16)vp.x, (_Float16)vp.y,
                   (_Float16)vp.z, (_Float16)vp.w };
        *(v4h*)&sA[row][c] = hb;
        *(v4h*)&sB[row][c] = hp;
    }

    // ---- Bias terms: threads 0..127 -> s_body rows, 128..255 -> s_pun rows ----
    {
        const float* src = (tid < BM) ? (bodyB + tid * H2) : (punB + (tid - BM) * H2);
        const float* w   = (tid < BM) ? w1 : w2;
        float s = 0.f;
#pragma unroll
        for (int d = 0; d < H2; d += 4) {
            float4 x  = *(const float4*)(src + d);
            float4 wv = *(const float4*)(w + d);
            s = fmaf(x.x, wv.x, s); s = fmaf(x.y, wv.y, s);
            s = fmaf(x.z, wv.z, s); s = fmaf(x.w, wv.w, s);
        }
        sBias[tid] = s;
    }

    __syncthreads();

    // ---- WMMA: each wave owns 32 rows x 64 cols (2x4 tiles of 16x16) ----
    const int waveM = wave & 3;     // 4 strips of 32 rows
    const int waveN = wave >> 2;    // 2 strips of 64 cols
    const int lr    = lane & 15;
    const int lh    = lane >> 4;    // 0 or 1 (half-wave)

    v8f acc[2][4];
#pragma unroll
    for (int mt = 0; mt < 2; ++mt)
#pragma unroll
        for (int nt = 0; nt < 4; ++nt)
            acc[mt][nt] = (v8f){};

#pragma unroll
    for (int kc = 0; kc < H2; kc += 32) {
        F16Frag a[2], bf[4];
        // A 16x32 f16 layout: lanes 0-15 hold K {0..7,16..23}, lanes 16-31 the rest
        const int aoff = kc + lh * 8;
#pragma unroll
        for (int mt = 0; mt < 2; ++mt) {
            const _Float16* p = &sA[waveM * 32 + mt * 16 + lr][aoff];
            a[mt].h[0] = *(const v8h*)p;
            a[mt].h[1] = *(const v8h*)(p + 16);
        }
        // B 32x16 f16 layout: lanes 0-15 hold K {0..7,16..23}, lanes 16-31 the rest
        const int boff = kc + lh * 8;
#pragma unroll
        for (int nt = 0; nt < 4; ++nt) {
            const _Float16* p = &sB[waveN * 64 + nt * 16 + lr][boff];
            bf[nt].h[0] = *(const v8h*)p;
            bf[nt].h[1] = *(const v8h*)(p + 16);
        }
#pragma unroll
        for (int mt = 0; mt < 2; ++mt)
#pragma unroll
            for (int nt = 0; nt < 4; ++nt)
                acc[mt][nt] = __builtin_amdgcn_wmma_f32_16x16x32_f16(
                    false, a[mt].v, false, bf[nt].v,
                    (short)0, acc[mt][nt], false, false);
    }

    // ---- Epilogue: add biases, store f32 ----
    // C/D layout: VGPR r -> row r (lanes 0-15) / row r+8 (lanes 16-31), col = lane%16
#pragma unroll
    for (int mt = 0; mt < 2; ++mt) {
        const int mBase = waveM * 32 + mt * 16;
#pragma unroll
        for (int nt = 0; nt < 4; ++nt) {
            const int col  = waveN * 64 + nt * 16 + lr;
            const float sp = sBias[BM + col];
#pragma unroll
            for (int r = 0; r < 8; ++r) {
                const int row = mBase + lh * 8 + r;
                const float val = acc[mt][nt][r] + sBias[row] + sp;
                out[((size_t)b * LSEQ + (i0 + row)) * LSEQ + (j0 + col)] = val;
            }
        }
    }
}

extern "C" void kernel_launch(void* const* d_in, const int* in_sizes, int n_in,
                              void* d_out, int out_size, void* d_ws, size_t ws_size,
                              hipStream_t stream) {
    // setup_inputs order: batch_size (int), body (f32), pun (f32), w_u (f32)
    const float* body = (const float*)d_in[1];
    const float* pun  = (const float*)d_in[2];
    const float* w_u  = (const float*)d_in[3];
    float* out = (float*)d_out;

    const int B = in_sizes[1] / (LSEQ * H2);   // 64
    dim3 grid(LSEQ / BN, LSEQ / BM, B);        // (8, 8, 64)
    alignment_matrix_wmma_kernel<<<grid, 256, 0, stream>>>(body, pun, w_u, out);
}
